// GRU_38732015075721
// MI455X (gfx1250) — matrix operand order repaired and checked by reading, not compiled, not run on hardware
//
#include <hip/hip_runtime.h>
#include <hip/hip_bf16.h>

typedef __attribute__((ext_vector_type(16))) _Float16 v16h;
typedef __attribute__((ext_vector_type(8)))  _Float16 v8h;
typedef __attribute__((ext_vector_type(8)))  float    v8f;

namespace {
constexpr int kB   = 512;   // batch
constexpr int kT   = 512;   // time steps
constexpr int kH   = 128;   // hidden
constexpr int kG   = 384;   // 3*H gate width
constexpr int kBM  = 16;    // batch rows per workgroup (one 16-row WMMA tile)
constexpr int kLDA = 136;   // padded halfs per row of A tiles in LDS
}

__device__ __forceinline__ float fast_sigmoid(float x) {
  return 1.0f / (1.0f + __expf(-x));
}
__device__ __forceinline__ float fast_tanh(float x) {
  // stable: exp overflow -> +inf -> 1; underflow -> 0 -> -1
  return 1.0f - 2.0f / (__expf(2.0f * x) + 1.0f);
}

// Low 32 bits of a generic pointer to __shared__ = LDS byte offset.
__device__ __forceinline__ unsigned lds_off(const void* p) {
  return (unsigned)(unsigned long long)p;
}
// CDNA5 async global->LDS copy (16B per lane), tracked by ASYNCcnt.
__device__ __forceinline__ void async_load_b128(unsigned lds_byte_off, const void* gaddr) {
  asm volatile("global_load_async_to_lds_b128 %0, %1, off"
               :: "v"(lds_byte_off), "v"(gaddr) : "memory");
}
__device__ __forceinline__ void wait_asynccnt0() {
  asm volatile("s_wait_asynccnt 0x0" ::: "memory");
}

// A fragment (16x32 f16, MxK) per CDNA5 ISA layout:
// lane: m = lane&15, hi = lane>>4; halfs j<8 : K = kt + hi*8 + j
//                                  halfs j>=8: K = kt + 16 + hi*8 + (j-8)
__device__ __forceinline__ v16h load_a_frag(const _Float16* s, int nl, int hi, int kt) {
  const _Float16* p = s + nl * kLDA + kt + hi * 8;
  v8h lo = *(const v8h*)(p);
  v8h hh = *(const v8h*)(p + 16);
  return __builtin_shufflevector(lo, hh, 0,1,2,3,4,5,6,7,8,9,10,11,12,13,14,15);
}
// Build a register-resident B fragment (32x16 f16, KxN) directly from the f32
// row-major weight W[384][128]: lane l holds column N = l&15 of W^T, element
// j is W[nt+(l&15)][kt+K], K = (l>>4)*8+j (j<8) or 16+(l>>4)*8+(j-8) (j>=8).
// One-time strided gather at kernel start; the fragment
// then lives in 8 VGPRs for the whole 512-step recurrence.
__device__ __forceinline__ v16h build_b_frag(const float* w, int lane, int kt, int nt) {
  v16h b;
#pragma unroll
  for (int j = 0; j < 16; ++j) b[j] = (_Float16)w[(size_t)(nt + (lane & 15)) * kH + kt + (lane >> 4) * 8 + (j < 8 ? j : 8 + j)];
  return b;
}

#define WMMA_F16F32(A, B, C) \
  __builtin_amdgcn_wmma_f32_16x16x32_f16(false, (A), false, (B), (short)0, (C), false, false)

// Hidden layers (input dim = H), in-place on seq[B][T][H] f16.
// Steady-state loop: async prefetch of x_{t+1}, 16 ds_load_b128 (A frags),
// 24 WMMA (all B frags register-resident), gate math, coalesced h writeback.
__global__ __launch_bounds__(256) void GRU_layerH_kernel(
    _Float16* seq,                       // [B][T][H] f16, read+write in place
    const float* __restrict__ wih,       // [384][128] f32
    const float* __restrict__ whh,       // [384][128] f32
    const float* __restrict__ bih,       // [384]
    const float* __restrict__ bhh)       // [384]
{
  __shared__ __align__(32) _Float16 sX16[2][kBM * kLDA];  // double-buffered x tile
  __shared__ __align__(32) _Float16 sH16[kBM * kLDA];     // h state f16 (WMMA A)
  __shared__ float sHf[kBM * kH];                         // h state f32 (exact carry)

  const int tid  = threadIdx.x;
  const int wave = tid >> 5;
  const int lane = tid & 31;
  const int nl   = lane & 15;
  const int hi   = lane >> 4;
  const long b0  = (long)blockIdx.x * kBM;

  for (int idx = tid; idx < kBM * kH; idx += 256) sHf[idx] = 0.0f;
  for (int idx = tid; idx < kBM * kLDA; idx += 256) sH16[idx] = (_Float16)0.0f;

  const int ntR = wave * 16, ntZ = kH + wave * 16, ntN = 2 * kH + wave * 16;
  const int col = wave * 16 + nl;
  const float brv  = bih[col] + bhh[col];
  const float bzv  = bih[kH + col] + bhh[kH + col];
  const float bxnv = bih[2 * kH + col];
  const float bhnv = bhh[2 * kH + col];

  // All 24 B fragments register-resident (192 VGPRs), built once.
  v16h bhhR[4], bhhZ[4], bhhN[4], bihR[4], bihZ[4], bihN[4];
#pragma unroll
  for (int kk = 0; kk < 4; ++kk) {
    const int kt = kk * 32;
    bhhR[kk] = build_b_frag(whh, lane, kt, ntR);
    bhhZ[kk] = build_b_frag(whh, lane, kt, ntZ);
    bhhN[kk] = build_b_frag(whh, lane, kt, ntN);
    bihR[kk] = build_b_frag(wih, lane, kt, ntR);
    bihZ[kk] = build_b_frag(wih, lane, kt, ntZ);
    bihN[kk] = build_b_frag(wih, lane, kt, ntN);
  }

  // Tile copy lane mapping: 256 threads x 16B == 16 rows x 128 halfs.
  const int pr = tid >> 4;          // row 0..15
  const int pc = (tid & 15) * 8;    // half-column 0,8,...,120

  // Preload x tile for t = 0.
  async_load_b128(lds_off(&sX16[0][pr * kLDA + pc]),
                  seq + ((b0 + pr) * kT + 0) * kH + pc);
  wait_asynccnt0();
  __syncthreads();

  for (int t = 0; t < kT; ++t) {
    // Prefetch next x tile while we compute (hidden behind WMMA + gating).
    if (t + 1 < kT) {
      async_load_b128(lds_off(&sX16[(t + 1) & 1][pr * kLDA + pc]),
                      seq + ((b0 + pr) * kT + (t + 1)) * kH + pc);
    }
    const _Float16* xb = sX16[t & 1];

    v8f accR = {}, accZ = {}, accXN = {}, accHN = {};
#pragma unroll
    for (int kk = 0; kk < 4; ++kk) {
      const int kt = kk * 32;
      v16h aX = load_a_frag(xb, nl, hi, kt);
      v16h aH = load_a_frag(sH16, nl, hi, kt);
      accR  = WMMA_F16F32(aX, bihR[kk], accR);
      accR  = WMMA_F16F32(aH, bhhR[kk], accR);
      accZ  = WMMA_F16F32(aX, bihZ[kk], accZ);
      accZ  = WMMA_F16F32(aH, bhhZ[kk], accZ);
      accXN = WMMA_F16F32(aX, bihN[kk], accXN);
      accHN = WMMA_F16F32(aH, bhhN[kk], accHN);
    }
    __syncthreads();  // all waves done reading sH16 / xb

    // Gates + h update; D layout: (m = 8*hi + rr, n = nl).
#pragma unroll
    for (int rr = 0; rr < 8; ++rr) {
      int m = hi * 8 + rr;
      float r  = fast_sigmoid(accR[rr] + brv);
      float z  = fast_sigmoid(accZ[rr] + bzv);
      float nn = fast_tanh(accXN[rr] + bxnv + r * (accHN[rr] + bhnv));
      float hp = sHf[m * kH + col];
      float hv = (1.0f - z) * nn + z * hp;
      sHf[m * kH + col]   = hv;
      sH16[m * kLDA + col] = (_Float16)hv;
    }
    __syncthreads();  // h writes visible to every wave

    // Coalesced in-place writeback of h_t (16B per thread).
    {
      v8h hv8 = *(const v8h*)&sH16[pr * kLDA + pc];
      *(v8h*)(seq + ((b0 + pr) * kT + t) * kH + pc) = hv8;
    }
    wait_asynccnt0();   // x_{t+1} tile resident in LDS
    __syncthreads();
  }
}

// First layer: input dim 5 -> xi via scalar FMAs; gh via WMMA. Output f16 seq.
__global__ __launch_bounds__(256) void GRU_layer0_kernel(
    const float* __restrict__ x,        // [B][T][5]
    const float* __restrict__ wih0,     // [384][5]
    const float* __restrict__ whh,      // [384][128]
    const float* __restrict__ bih,
    const float* __restrict__ bhh,
    _Float16* __restrict__ out)         // [B][T][H] f16
{
  __shared__ __align__(32) _Float16 sH16[kBM * kLDA];
  __shared__ float sHf[kBM * kH];
  __shared__ float sW0[kG * 5];
  __shared__ float sX0[kBM * 5];

  const int tid  = threadIdx.x;
  const int wave = tid >> 5;
  const int lane = tid & 31;
  const int nl   = lane & 15;
  const int hi   = lane >> 4;
  const long b0  = (long)blockIdx.x * kBM;

  for (int idx = tid; idx < kG * 5; idx += 256) sW0[idx] = wih0[idx];
  for (int idx = tid; idx < kBM * kH; idx += 256) sHf[idx] = 0.0f;
  for (int idx = tid; idx < kBM * kLDA; idx += 256) sH16[idx] = (_Float16)0.0f;
  __syncthreads();

  const int ntR = wave * 16, ntZ = kH + wave * 16, ntN = 2 * kH + wave * 16;
  const int col = wave * 16 + nl;
  const float brv  = bih[col] + bhh[col];
  const float bzv  = bih[kH + col] + bhh[kH + col];
  const float bxnv = bih[2 * kH + col];
  const float bhnv = bhh[2 * kH + col];

  v16h bhhR[4], bhhZ[4], bhhN[4];
#pragma unroll
  for (int kk = 0; kk < 4; ++kk) {
    const int kt = kk * 32;
    bhhR[kk] = build_b_frag(whh, lane, kt, ntR);
    bhhZ[kk] = build_b_frag(whh, lane, kt, ntZ);
    bhhN[kk] = build_b_frag(whh, lane, kt, ntN);
  }
  float wr[5], wz[5], wn[5];
#pragma unroll
  for (int i = 0; i < 5; ++i) {
    wr[i] = sW0[col * 5 + i];
    wz[i] = sW0[(kH + col) * 5 + i];
    wn[i] = sW0[(2 * kH + col) * 5 + i];
  }

  const int pr = tid >> 4;
  const int pc = (tid & 15) * 8;

  for (int t = 0; t < kT; ++t) {
    if (tid < kBM * 5) {
      int r = tid / 5, c = tid % 5;
      sX0[tid] = x[((b0 + r) * kT + t) * 5 + c];
    }
    __syncthreads();

    v8f accR = {}, accZ = {}, accHN = {};
#pragma unroll
    for (int kk = 0; kk < 4; ++kk) {
      v16h aH = load_a_frag(sH16, nl, hi, kk * 32);
      accR  = WMMA_F16F32(aH, bhhR[kk], accR);
      accZ  = WMMA_F16F32(aH, bhhZ[kk], accZ);
      accHN = WMMA_F16F32(aH, bhhN[kk], accHN);
    }
    __syncthreads();

#pragma unroll
    for (int rr = 0; rr < 8; ++rr) {
      int m = hi * 8 + rr;
      float xiR = 0.0f, xiZ = 0.0f, xiN = 0.0f;
#pragma unroll
      for (int i = 0; i < 5; ++i) {
        float xv = sX0[m * 5 + i];
        xiR = fmaf(xv, wr[i], xiR);
        xiZ = fmaf(xv, wz[i], xiZ);
        xiN = fmaf(xv, wn[i], xiN);
      }
      float r  = fast_sigmoid(accR[rr] + xiR + brv);
      float z  = fast_sigmoid(accZ[rr] + xiZ + bzv);
      float nn = fast_tanh(xiN + bxnv + r * (accHN[rr] + bhnv));
      float hp = sHf[m * kH + col];
      float hv = (1.0f - z) * nn + z * hp;
      sHf[m * kH + col]   = hv;
      sH16[m * kLDA + col] = (_Float16)hv;
    }
    __syncthreads();

    {
      v8h hv8 = *(const v8h*)&sH16[pr * kLDA + pc];
      *(v8h*)(out + ((b0 + pr) * kT + t) * kH + pc) = hv8;
    }
    __syncthreads();
  }
}

// Final FC: out[b] = dot(h_last[b], fc_w) + fc_b, h in f16.
__global__ void GRU_fc_kernel(const _Float16* __restrict__ seq,
                              const float* __restrict__ fcw,
                              const float* __restrict__ fcb,
                              float* __restrict__ outp) {
  int b = blockIdx.x * blockDim.x + threadIdx.x;
  if (b < kB) {
    const _Float16* h = seq + ((size_t)b * kT + (kT - 1)) * kH;
    float acc = fcb[0];
#pragma unroll 8
    for (int k = 0; k < kH; ++k) acc = fmaf((float)h[k], fcw[k], acc);
    outp[b] = acc;
  }
}

extern "C" void kernel_launch(void* const* d_in, const int* in_sizes, int n_in,
                              void* d_out, int out_size, void* d_ws, size_t ws_size,
                              hipStream_t stream) {
  const float* x        = (const float*)d_in[0];
  const float* wih0     = (const float*)d_in[1];
  const float* whh0     = (const float*)d_in[2];
  const float* bih0     = (const float*)d_in[3];
  const float* bhh0     = (const float*)d_in[4];
  const float* wih_rest = (const float*)d_in[5];   // [3][384][128]
  const float* whh_rest = (const float*)d_in[6];   // [3][384][128]
  const float* bih_rest = (const float*)d_in[7];   // [3][384]
  const float* bhh_rest = (const float*)d_in[8];   // [3][384]
  const float* fcw      = (const float*)d_in[9];
  const float* fcb      = (const float*)d_in[10];

  // Workspace: one f16 sequence buffer [B][T][H] (~67 MB); layers 1-3 in place.
  _Float16* seq = (_Float16*)d_ws;

  GRU_layer0_kernel<<<dim3(kB / kBM), dim3(256), 0, stream>>>(
      x, wih0, whh0, bih0, bhh0, seq);

  for (int l = 0; l < 3; ++l) {
    GRU_layerH_kernel<<<dim3(kB / kBM), dim3(256), 0, stream>>>(
        seq, wih_rest + (size_t)l * kG * kH, whh_rest + (size_t)l * kG * kH,
        bih_rest + (size_t)l * kG, bhh_rest + (size_t)l * kG);
  }

  GRU_fc_kernel<<<dim3(2), dim3(256), 0, stream>>>(seq, fcw, fcb, (float*)d_out);
}
